// ContrastiveLoss_37194416783670
// MI455X (gfx1250) — matrix operand order repaired and checked by reading, not compiled, not run on hardware
//
#include <hip/hip_runtime.h>

typedef _Float16 v16h  __attribute__((ext_vector_type(16)));
typedef _Float16 v8h   __attribute__((ext_vector_type(8)));
typedef float    v8f   __attribute__((ext_vector_type(8)));
typedef int      v4i   __attribute__((ext_vector_type(4)));

#define NROWS 8192
#define DDIM  256
#define TEMP_INV 10.0f

#define ROWS_PER_BLOCK 128          // 8 waves x 16-row stripes
#define COLS_PER_BLOCK 1024         // 8 column-groups cover 8192
#define TILE_COLS      32           // pair of 16-col WMMA tiles per iteration
#define NITER          (COLS_PER_BLOCK / TILE_COLS)   // 32
#define LDS_COL_F16    264          // 256 f16 payload + 8 f16 pad (528 B) -> bank-conflict-free
#define BUF_F16        (TILE_COLS * LDS_COL_F16)      // 8448 f16 = 16896 B per buffer

// --- CDNA5 async global->LDS copy (ASYNCcnt), guarded with sync fallback. ---
// Probe-discovered signature: (int4 addrspace(1)*, int4 addrspace(3)*, imm, imm)
#if defined(__HIP_DEVICE_COMPILE__) && __has_builtin(__builtin_amdgcn_global_load_async_to_lds_b128)
#define ASYNC_CP16(g, l, OFF)                                                   \
    __builtin_amdgcn_global_load_async_to_lds_b128(                             \
        (__attribute__((address_space(1))) v4i*)((const char*)(g) + (OFF)),     \
        (__attribute__((address_space(3))) v4i*)((char*)(l) + (OFF)), 0, 0)
#else
#define ASYNC_CP16(g, l, OFF)                                                   \
    (*(v8h*)((char*)(l) + (OFF)) = *(const v8h*)((const char*)(g) + (OFF)))
#endif

__device__ __forceinline__ void wait_async0() {
#if defined(__HIP_DEVICE_COMPILE__) && __has_builtin(__builtin_amdgcn_s_wait_asynccnt)
    __builtin_amdgcn_s_wait_asynccnt(0);
#elif defined(__HIP_DEVICE_COMPILE__)
    asm volatile("s_wait_asynccnt 0x0" ::: "memory");
#endif
}

// Cooperative stage of one 32-column tile (16.5 KB) into LDS: thread t copies
// 64 B of column c = t>>3, chunk j = t&7, as 4 x b128 async transfers.
__device__ __forceinline__ void stage_tile(const _Float16* __restrict__ rH,
                                           _Float16* __restrict__ sdst,
                                           int colbase, int t) {
    const int c = t >> 3;
    const int j = t & 7;
    const _Float16* g = rH + (size_t)(colbase + c) * DDIM + j * 32;
    _Float16*       l = sdst + c * LDS_COL_F16 + j * 32;
    ASYNC_CP16(g, l, 0);
    ASYNC_CP16(g, l, 16);
    ASYNC_CP16(g, l, 32);
    ASYNC_CP16(g, l, 48);
}

// ---------------------------------------------------------------------------
// Kernel 1: L2-normalize rows (torch F.normalize semantics), emit f16.
// ---------------------------------------------------------------------------
__global__ void __launch_bounds__(256) normalize_rows_f16(
    const float* __restrict__ x, _Float16* __restrict__ rH)
{
    __shared__ float red[256];
    const int row = blockIdx.x;
    const int t   = threadIdx.x;
    const float v = x[row * DDIM + t];
    red[t] = v * v;
    __syncthreads();
    for (int s = 128; s > 0; s >>= 1) {
        if (t < s) red[t] += red[t + s];
        __syncthreads();
    }
    const float nrm   = sqrtf(red[0]);
    const float scale = 1.0f / fmaxf(nrm, 1e-12f);
    rH[row * DDIM + t] = (_Float16)(v * scale);
}

// ---------------------------------------------------------------------------
// Kernel 2: fused sim = r@r^T -> masked exp -> row sums, 2-level blocked.
// Grid (8,64): block = 128 rows x 1024 cols.  Wave w owns rows w*16..+15 with
// A resident in VGPRs; all 8 waves share async-staged, double-buffered B tiles
// in LDS (B fetched once per block instead of once per wave: 8x less L2).
// ---------------------------------------------------------------------------
__global__ void __launch_bounds__(256) contrastive_rowsum(
    const _Float16* __restrict__ rH, const int* __restrict__ lab,
    float* __restrict__ rowsumG)
{
    __shared__ _Float16 sbuf[2 * BUF_F16];   // ~33 KB double buffer

    const int tid  = threadIdx.x;
    const int wave = tid >> 5;
    const int lane = tid & 31;
    const int half = lane >> 4;   // 0: lanes 0-15, 1: lanes 16-31
    const int l16  = lane & 15;

    const int row_base  = blockIdx.y * ROWS_PER_BLOCK + wave * 16;
    const int col_group = blockIdx.x * COLS_PER_BLOCK;

    // --- A fragments (16 rows x K=256), resident for the whole sweep.
    // Lane holds row (lane%16); K = kc*32 + half*8 + {0..7} and +16.
    v16h A[8];
    const _Float16* arow = rH + (size_t)(row_base + l16) * DDIM;
#pragma unroll
    for (int kc = 0; kc < 8; ++kc) {
        const _Float16* p = arow + kc * 32 + half * 8;
        v8h lo = *(const v8h*)(p);
        v8h hi = *(const v8h*)(p + 16);
        A[kc] = __builtin_shufflevector(lo, hi,
                    0,1,2,3,4,5,6,7,8,9,10,11,12,13,14,15);
    }

    int labr[8];
#pragma unroll
    for (int v = 0; v < 8; ++v) labr[v] = lab[row_base + half * 8 + v];

    float rowacc[8];
#pragma unroll
    for (int v = 0; v < 8; ++v) rowacc[v] = 0.0f;

    // Prologue: stage tile 0 into buffer 0.
    stage_tile(rH, sbuf, col_group, tid);
    wait_async0();
    __syncthreads();

    for (int it = 0; it < NITER; ++it) {
        // Stage next tile into the other buffer while computing this one.
        if (it + 1 < NITER)
            stage_tile(rH, sbuf + ((it + 1) & 1) * BUF_F16,
                       col_group + (it + 1) * TILE_COLS, tid);

        const _Float16* bp0 = sbuf + (it & 1) * BUF_F16 + l16 * LDS_COL_F16 + half * 8;
        const _Float16* bp1 = bp0 + 16 * LDS_COL_F16;
        const int colbase = col_group + it * TILE_COLS;
        const int lcol0 = lab[colbase + l16];
        const int lcol1 = lab[colbase + 16 + l16];

        // Two independent WMMA chains over K=256 (8 chunks of 32), B from LDS.
        v8f c0 = {}, c1 = {};
#pragma unroll
        for (int kc = 0; kc < 8; ++kc) {
            v8h lo0 = *(const v8h*)(bp0 + kc * 32);
            v8h hi0 = *(const v8h*)(bp0 + kc * 32 + 16);
            v8h lo1 = *(const v8h*)(bp1 + kc * 32);
            v8h hi1 = *(const v8h*)(bp1 + kc * 32 + 16);
            v16h b0 = __builtin_shufflevector(lo0, hi0,
                          0,1,2,3,4,5,6,7,8,9,10,11,12,13,14,15);
            v16h b1 = __builtin_shufflevector(lo1, hi1,
                          0,1,2,3,4,5,6,7,8,9,10,11,12,13,14,15);
            c0 = __builtin_amdgcn_wmma_f32_16x16x32_f16(
                    false, A[kc], false, b0, (short)0, c0, false, false);
            c1 = __builtin_amdgcn_wmma_f32_16x16x32_f16(
                    false, A[kc], false, b1, (short)0, c1, false, false);
        }

        // Same-label entries (incl. diag) contribute exp(0)=1; different-label
        // contribute exp(sim/T).  Matches the reference denominator exactly.
#pragma unroll
        for (int v = 0; v < 8; ++v) {
            rowacc[v] += (labr[v] == lcol0) ? 1.0f : __expf(c0[v] * TEMP_INV);
            rowacc[v] += (labr[v] == lcol1) ? 1.0f : __expf(c1[v] * TEMP_INV);
        }

        wait_async0();      // next tile landed in LDS (own wave's share)
        __syncthreads();    // all waves' shares landed
    }

    // --- Reduce across the 16 lanes sharing each C-row (masks < 16 keep the
    // butterfly inside each 16-lane half of the wave32).
#pragma unroll
    for (int v = 0; v < 8; ++v) {
        float s = rowacc[v];
        s += __shfl_xor(s, 8, 32);
        s += __shfl_xor(s, 4, 32);
        s += __shfl_xor(s, 2, 32);
        s += __shfl_xor(s, 1, 32);
        rowacc[v] = s;
    }

    // Rows are unique per wave; accumulate across column-group blocks globally.
    if (l16 == 0) {
#pragma unroll
        for (int v = 0; v < 8; ++v)
            atomicAdd(&rowsumG[row_base + half * 8 + v], rowacc[v]);
    }
}

// ---------------------------------------------------------------------------
// Init / finalize kernels (graph-capture safe).
// ---------------------------------------------------------------------------
__global__ void zero_rowsum(float* rs) {
    rs[blockIdx.x * 256 + threadIdx.x] = 0.0f;
}

__global__ void __launch_bounds__(256) finalize_loss(
    const float* __restrict__ rs, float* __restrict__ out)
{
    __shared__ float red[256];
    float a = 0.0f;
    for (int r = threadIdx.x; r < NROWS; r += 256)
        a += logf(rs[r] + 1.0f + 1e-8f);   // den = rowsum + num_diag(=1), +eps
    red[threadIdx.x] = a;
    __syncthreads();
    for (int s = 128; s > 0; s >>= 1) {
        if (threadIdx.x < s) red[threadIdx.x] += red[threadIdx.x + s];
        __syncthreads();
    }
    if (threadIdx.x == 0) out[0] = red[0] * (1.0f / (float)NROWS);
}

extern "C" void kernel_launch(void* const* d_in, const int* in_sizes, int n_in,
                              void* d_out, int out_size, void* d_ws, size_t ws_size,
                              hipStream_t stream) {
    (void)in_sizes; (void)n_in; (void)out_size; (void)ws_size;
    const float* x   = (const float*)d_in[0];   // representations [8192,256] f32
    const int*   lab = (const int*)d_in[1];     // pseudo_labels   [8192]
    float*       out = (float*)d_out;           // scalar loss f32

    // Workspace: 4 MB normalized f16 matrix + 32 KB per-row sums.
    _Float16* rH      = (_Float16*)d_ws;
    float*    rowsumG = (float*)((char*)d_ws + (size_t)NROWS * DDIM * sizeof(_Float16));

    zero_rowsum<<<NROWS / 256, 256, 0, stream>>>(rowsumG);
    normalize_rows_f16<<<NROWS, 256, 0, stream>>>(x, rH);
    contrastive_rowsum<<<dim3(NROWS / COLS_PER_BLOCK, NROWS / ROWS_PER_BLOCK),
                         256, 0, stream>>>(rH, lab, rowsumG);
    finalize_loss<<<1, 256, 0, stream>>>(rowsumG, out);
}
